// ConvFlow_23545010716749
// MI455X (gfx1250) — matrix operand order repaired and checked by reading, not compiled, not run on hardware
//
#include <hip/hip_runtime.h>
#include <hip/hip_bf16.h>
#include <math.h>

// ---------------------------------------------------------------------------
// ConvFlow for MI455X (gfx1250, wave32, WMMA).
// Layouts:
//   h   : f32  [c][n]   c in [0,192), n = b*T + t   (192 MB, in d_ws)
//   y16 : f16  [n][c]   WMMA-B friendly (16 contiguous halves per K-chunk)
//   w16 : f16  [o][c]   per-layer 192x192 pointwise weights
//   wp16: f16  [o][c]   projection weights padded 29 -> 32 rows
// ---------------------------------------------------------------------------

#define BATCH 128
#define TLEN  2048
#define NTOT  (BATCH * TLEN)     // 262144 positions
#define FCH   192
#define NPROJ 29
#define NPROJP 32                // padded to 2 M-tiles of 16

typedef _Float16 v16h __attribute__((ext_vector_type(16)));
typedef _Float16 v8h  __attribute__((ext_vector_type(8)));
typedef float    v8f  __attribute__((ext_vector_type(8)));

static __device__ __forceinline__ float gelu_exact(float v) {
    return 0.5f * v * (1.0f + erff(v * 0.70710678118654752f));
}
static __device__ __forceinline__ float softplus_f(float v) {
    return (v > 20.0f) ? v : log1pf(expf(v));
}
static __device__ __forceinline__ v16h combine16(v8h lo, v8h hi) {
    v16h r;
#pragma unroll
    for (int e = 0; e < 8; ++e) { r[e] = lo[e]; r[e + 8] = hi[e]; }
    return r;
}

// ---------------------------------------------------------------------------
// K0: weights -> f16 (pointwise layers + padded projection)
// ---------------------------------------------------------------------------
__global__ __launch_bounds__(256) void k0_prep_weights(
    const float* __restrict__ pw_w, const float* __restrict__ proj_w,
    _Float16* __restrict__ w16, _Float16* __restrict__ wp16)
{
    int e = blockIdx.x * 256 + threadIdx.x;
    if (e < 3 * FCH * FCH) w16[e] = (_Float16)pw_w[e];
    if (e < NPROJP * FCH) {
        int o = e / FCH, c = e % FCH;
        wp16[e] = (_Float16)((o < NPROJ) ? proj_w[o * FCH + c] : 0.0f);
    }
}

// ---------------------------------------------------------------------------
// K1: pre-projection h[c][n] = pre_w[c]*x0[n] + pre_b[c]; also emit x0 half
// ---------------------------------------------------------------------------
__global__ __launch_bounds__(256) void k1_pre(
    const float* __restrict__ x, const float* __restrict__ mask,
    const float* __restrict__ pw, const float* __restrict__ pb,
    float* __restrict__ h, float* __restrict__ out)
{
    int n = blockIdx.x * 256 + threadIdx.x;
    int b = n >> 11, t = n & (TLEN - 1);
    float x0 = x[(size_t)b * 2 * TLEN + t];
    out[(size_t)b * 2 * TLEN + t] = x0 * mask[n];
#pragma unroll 4
    for (int c = 0; c < FCH; ++c)
        h[(size_t)c * NTOT + n] = pw[c] * x0 + pb[c];
}

// ---------------------------------------------------------------------------
// K2: depthwise conv (K=3, dil) on masked h, + LayerNorm(ch) + GELU
//     -> y16 [n][c] f16.  64 positions per block, LDS-staged.
// ---------------------------------------------------------------------------
__global__ __launch_bounds__(256) void k2_dwconv_ln_gelu(
    const float* __restrict__ h, const float* __restrict__ mask,
    const float* __restrict__ sw, const float* __restrict__ sb,
    const float* __restrict__ g, const float* __restrict__ bb,
    _Float16* __restrict__ y16, int dil)
{
    __shared__ float s[FCH * 65];          // padded stride: conflict-free
    __shared__ float smean[64], srstd[64];
    int n0 = blockIdx.x * 64;

    for (int e = threadIdx.x; e < FCH * 64; e += 256) {
        int c = e >> 6, nl = e & 63;
        int n = n0 + nl, t = n & (TLEN - 1), b = n >> 11;
        float w0 = sw[c * 3 + 0], w1 = sw[c * 3 + 1], w2 = sw[c * 3 + 2];
        float acc = sb[c];
        size_t base = (size_t)c * NTOT + (size_t)b * TLEN;
        int tm = t - dil, tp = t + dil;
        if (tm >= 0)   acc += w0 * h[base + tm] * mask[b * TLEN + tm];
        acc += w1 * h[base + t] * mask[n];
        if (tp < TLEN) acc += w2 * h[base + tp] * mask[b * TLEN + tp];
        s[c * 65 + nl] = acc;
    }
    __syncthreads();
    if (threadIdx.x < 64) {
        int p = threadIdx.x;
        float sm = 0.f, sq = 0.f;
        for (int c = 0; c < FCH; ++c) { float v = s[c * 65 + p]; sm += v; sq += v * v; }
        float mean = sm * (1.0f / FCH);
        float var  = sq * (1.0f / FCH) - mean * mean;
        smean[p] = mean; srstd[p] = rsqrtf(var + 1e-5f);
    }
    __syncthreads();
    for (int e = threadIdx.x; e < FCH * 64; e += 256) {
        int c = e % FCH, nl = e / FCH;
        float v = s[c * 65 + nl];
        float o = (v - smean[nl]) * srstd[nl] * g[c] + bb[c];
        y16[(size_t)(n0 + nl) * FCH + c] = (_Float16)gelu_exact(o);
    }
}

// ---------------------------------------------------------------------------
// K3: pointwise 192x192 matmul (WMMA f16->f32) + bias + LayerNorm + GELU
//     + residual into h.  8 waves/block, 16 columns/wave, 12 M-tiles,
//     K=192 = 6 chunks of 32.  LN done in registers via shfl_xor(16).
//     Optionally emits masked h as f16 [n][c] (last layer, feeds K4).
// ---------------------------------------------------------------------------
__global__ __launch_bounds__(256) void k3_pointwise_wmma(
    const _Float16* __restrict__ W, const float* __restrict__ bias,
    const float* __restrict__ g, const float* __restrict__ bb,
    float* __restrict__ h, const _Float16* __restrict__ y,
    _Float16* __restrict__ h16, const float* __restrict__ mask)
{
    int lane = threadIdx.x & 31;
    int wave = threadIdx.x >> 5;
    int hi   = lane >> 4;                     // 0 | 1  (lane half)
    int n    = blockIdx.x * 128 + wave * 16 + (lane & 15);

    // B fragments: Y column n, 6 K-chunks, 16 contiguous halves each
    const _Float16* yb = y + (size_t)n * FCH;
    v16h bf[6];
#pragma unroll
    for (int kc = 0; kc < 6; ++kc) {
        int off = kc * 32 + hi * 16;
        bf[kc] = combine16(*(const v8h*)(yb + off), *(const v8h*)(yb + off + 8));
    }

    v8f acc[12];
#pragma unroll
    for (int mt = 0; mt < 12; ++mt) {
        v8f c; 
#pragma unroll
        for (int e = 0; e < 8; ++e) c[e] = 0.0f;
        const _Float16* wrow = W + (size_t)(mt * 16 + (lane & 15)) * FCH;
#pragma unroll
        for (int kc = 0; kc < 6; ++kc) {
            int off = kc * 32 + hi * 16;      // A layout: halves [k..k+15] contiguous
            v16h a = combine16(*(const v8h*)(wrow + off),
                               *(const v8h*)(wrow + off + 8));
            c = __builtin_amdgcn_wmma_f32_16x16x32_f16(
                    false, a, false, bf[kc], (short)0, c, false, false);
        }
        acc[mt] = c;
    }

    // bias + LayerNorm over 192 channels (this lane + lane^16 hold them all)
    float sum = 0.f, sq = 0.f;
#pragma unroll
    for (int mt = 0; mt < 12; ++mt)
#pragma unroll
        for (int r = 0; r < 8; ++r) {
            int m = mt * 16 + r + hi * 8;
            float t = acc[mt][r] + bias[m];
            sum += t; sq += t * t;
        }
    sum += __shfl_xor(sum, 16, 32);
    sq  += __shfl_xor(sq, 16, 32);
    float mean = sum * (1.0f / FCH);
    float var  = sq * (1.0f / FCH) - mean * mean;
    float rs   = rsqrtf(var + 1e-5f);
    float mk   = (h16 != nullptr) ? mask[n] : 1.0f;

#pragma unroll
    for (int mt = 0; mt < 12; ++mt)
#pragma unroll
        for (int r = 0; r < 8; ++r) {
            int m = mt * 16 + r + hi * 8;
            float t = acc[mt][r] + bias[m];
            t = (t - mean) * rs * g[m] + bb[m];
            t = gelu_exact(t);
            size_t idx = (size_t)m * NTOT + n;
            float hn = h[idx] + t;
            h[idx] = hn;
            if (h16 != nullptr) h16[(size_t)n * FCH + m] = (_Float16)(hn * mk);
        }
}

// ---------------------------------------------------------------------------
// K4: projection (32x192 WMMA, rows >=29 zero-padded) + rational quadratic
//     spline on x1 + per-wave logdet partials (deterministic).
// ---------------------------------------------------------------------------
__global__ __launch_bounds__(256) void k4_proj_rqs(
    const _Float16* __restrict__ Wp, const float* __restrict__ pbias,
    const _Float16* __restrict__ h16, const float* __restrict__ x,
    const float* __restrict__ mask, float* __restrict__ out,
    float* __restrict__ partial)
{
    int lane = threadIdx.x & 31;
    int wave = threadIdx.x >> 5;
    int hi   = lane >> 4;
    int n0   = blockIdx.x * 128 + wave * 16;
    int n    = n0 + (lane & 15);

    const _Float16* yb = h16 + (size_t)n * FCH;
    v16h bf[6];
#pragma unroll
    for (int kc = 0; kc < 6; ++kc) {
        int off = kc * 32 + hi * 16;
        bf[kc] = combine16(*(const v8h*)(yb + off), *(const v8h*)(yb + off + 8));
    }
    v8f acc[2];
#pragma unroll
    for (int mt = 0; mt < 2; ++mt) {
        v8f c;
#pragma unroll
        for (int e = 0; e < 8; ++e) c[e] = 0.0f;
        const _Float16* wrow = Wp + (size_t)(mt * 16 + (lane & 15)) * FCH;
#pragma unroll
        for (int kc = 0; kc < 6; ++kc) {
            int off = kc * 32 + hi * 16;
            v16h a = combine16(*(const v8h*)(wrow + off),
                               *(const v8h*)(wrow + off + 8));
            c = __builtin_amdgcn_wmma_f32_16x16x32_f16(
                    false, a, false, bf[kc], (short)0, c, false, false);
        }
        acc[mt] = c;
    }

    // own rows: tile0 -> o = r + 8*hi ; tile1 -> o = 16 + r + 8*hi
    float own0[8], own1[8], oth0[8], oth1[8];
#pragma unroll
    for (int r = 0; r < 8; ++r) {
        int o0 = r + 8 * hi, o1 = 16 + r + 8 * hi;
        own0[r] = acc[0][r] + ((o0 < NPROJ) ? pbias[o0] : 0.0f);
        own1[r] = acc[1][r] + ((o1 < NPROJ) ? pbias[o1] : 0.0f);
        oth0[r] = __shfl_xor(own0[r], 16, 32);
        oth1[r] = __shfl_xor(own1[r], 16, 32);
    }
    float full[32];
#pragma unroll
    for (int r = 0; r < 8; ++r) {
        full[r]      = hi ? oth0[r] : own0[r];
        full[r + 8]  = hi ? own0[r] : oth0[r];
        full[r + 16] = hi ? oth1[r] : own1[r];
        full[r + 24] = hi ? own1[r] : oth1[r];
    }

    // ---------------- rational quadratic spline ----------------
    const float tb = 5.0f;
    const float scale = 0.07216878364870323f;     // 1/sqrt(192)
    int b = n >> 11, t = n & (TLEN - 1);
    float x1 = x[(size_t)b * 2 * TLEN + TLEN + t];
    float mkv = mask[n];

    float uw[10], uh[10], derivs[11];
#pragma unroll
    for (int k = 0; k < 10; ++k) { uw[k] = full[k] * scale; uh[k] = full[10 + k] * scale; }
    // pad value: MIN_D + softplus(log(expm1(1-MIN_D))) == 1.0
    derivs[0] = 1.0f; derivs[10] = 1.0f;
#pragma unroll
    for (int k = 1; k < 10; ++k) derivs[k] = 0.001f + softplus_f(full[20 + k - 1]);

    // softmax -> affine -> cumsum for widths / heights
    float cw[11], widths[10], ch[11], heights[10];
    {
        float mx = uw[0];
#pragma unroll
        for (int k = 1; k < 10; ++k) mx = fmaxf(mx, uw[k]);
        float ex[10], s = 0.f;
#pragma unroll
        for (int k = 0; k < 10; ++k) { ex[k] = expf(uw[k] - mx); s += ex[k]; }
        float inv = 1.0f / s, run = 0.f;
        cw[0] = -tb;
#pragma unroll
        for (int k = 0; k < 10; ++k) {
            float w = 0.001f + 0.99f * ex[k] * inv;
            run += w;
            cw[k + 1] = 2.0f * tb * run - tb;
        }
        cw[10] = tb;
#pragma unroll
        for (int k = 0; k < 10; ++k) widths[k] = cw[k + 1] - cw[k];
    }
    {
        float mx = uh[0];
#pragma unroll
        for (int k = 1; k < 10; ++k) mx = fmaxf(mx, uh[k]);
        float ex[10], s = 0.f;
#pragma unroll
        for (int k = 0; k < 10; ++k) { ex[k] = expf(uh[k] - mx); s += ex[k]; }
        float inv = 1.0f / s, run = 0.f;
        ch[0] = -tb;
#pragma unroll
        for (int k = 0; k < 10; ++k) {
            float w = 0.001f + 0.99f * ex[k] * inv;
            run += w;
            ch[k + 1] = 2.0f * tb * run - tb;
        }
        ch[10] = tb;
#pragma unroll
        for (int k = 0; k < 10; ++k) heights[k] = ch[k + 1] - ch[k];
    }

    float xc = fminf(fmaxf(x1, -tb), tb);
    int cnt = 0;
#pragma unroll
    for (int k = 0; k < 11; ++k) {
        float loc = (k == 10) ? (tb + 1e-6f) : cw[k];
        cnt += (xc >= loc) ? 1 : 0;
    }
    int idx = min(max(cnt - 1, 0), 9);

    float in_cw = 0.f, in_w = 1.f, in_ch = 0.f, in_h = 1.f, delta = 1.f, d0 = 1.f;
#pragma unroll
    for (int k = 0; k < 10; ++k) {
        bool sel = (k == idx);
        in_cw = sel ? cw[k]      : in_cw;
        in_w  = sel ? widths[k]  : in_w;
        in_ch = sel ? ch[k]      : in_ch;
        in_h  = sel ? heights[k] : in_h;
        delta = sel ? (heights[k] / widths[k]) : delta;
        d0    = sel ? derivs[k]  : d0;
    }
    float d1 = 1.f;
#pragma unroll
    for (int k = 1; k < 11; ++k) d1 = (k == idx + 1) ? derivs[k] : d1;

    float theta = (xc - in_cw) / in_w;
    float t1    = theta * (1.0f - theta);
    float denom = delta + (d0 + d1 - 2.0f * delta) * t1;
    float num   = in_h * (delta * theta * theta + d0 * t1);
    float outv  = in_ch + num / denom;
    float ld    = 2.0f * logf(delta)
                + logf(d1 * theta * theta + 2.0f * delta * t1 + d0 * (1.0f - theta) * (1.0f - theta))
                - 2.0f * logf(denom);

    bool inside = (x1 >= -tb) && (x1 <= tb);
    float res = inside ? outv : x1;
    ld        = inside ? ld : 0.0f;

    if (lane < 16)
        out[(size_t)b * 2 * TLEN + TLEN + t] = res * mkv;

    // deterministic logdet partial: lanes 0-15 and 16-31 computed identical
    // values, full-wave sum counts each position twice -> scale by 0.5
    float w = ld * mkv;
#pragma unroll
    for (int s = 16; s > 0; s >>= 1) w += __shfl_xor(w, s, 32);
    if (lane == 0)
        partial[b * 128 + ((n0 & (TLEN - 1)) >> 4)] = w * 0.5f;
}

// ---------------------------------------------------------------------------
// K5: logdet[b] = sum over 128 wave partials (fixed order, deterministic)
// ---------------------------------------------------------------------------
__global__ __launch_bounds__(128) void k5_logdet_reduce(
    const float* __restrict__ partial, float* __restrict__ logdet)
{
    int b = threadIdx.x;
    float s = 0.f;
    for (int k = 0; k < 128; ++k) s += partial[b * 128 + k];
    logdet[b] = s;
}

// ---------------------------------------------------------------------------
extern "C" void kernel_launch(void* const* d_in, const int* in_sizes, int n_in,
                              void* d_out, int out_size, void* d_ws, size_t ws_size,
                              hipStream_t stream) {
    (void)in_sizes; (void)n_in; (void)out_size; (void)ws_size;

    const float* x      = (const float*)d_in[0];
    const float* mask   = (const float*)d_in[1];
    const float* pre_w  = (const float*)d_in[2];
    const float* pre_b  = (const float*)d_in[3];
    const float* sep_w  = (const float*)d_in[4];
    const float* sep_b  = (const float*)d_in[5];
    const float* pw_w   = (const float*)d_in[6];
    const float* pw_b   = (const float*)d_in[7];
    const float* n1_g   = (const float*)d_in[8];
    const float* n1_b   = (const float*)d_in[9];
    const float* n2_g   = (const float*)d_in[10];
    const float* n2_b   = (const float*)d_in[11];
    const float* proj_w = (const float*)d_in[12];
    const float* proj_b = (const float*)d_in[13];

    float* out    = (float*)d_out;
    float* logdet = out + (size_t)BATCH * 2 * TLEN;

    // workspace carve-up
    float*    h    = (float*)d_ws;                       // FCH*NTOT f32
    _Float16* y16  = (_Float16*)(h + (size_t)FCH * NTOT);// FCH*NTOT f16 (also h16)
    _Float16* w16  = y16 + (size_t)FCH * NTOT;           // 3*192*192
    _Float16* wp16 = w16 + 3 * FCH * FCH;                // 32*192
    float*    part = (float*)(wp16 + NPROJP * FCH);      // 128*128

    k0_prep_weights<<<(3 * FCH * FCH + 255) / 256, 256, 0, stream>>>(pw_w, proj_w, w16, wp16);
    k1_pre<<<NTOT / 256, 256, 0, stream>>>(x, mask, pre_w, pre_b, h, out);

    int dils[3] = {1, 3, 9};
    for (int i = 0; i < 3; ++i) {
        k2_dwconv_ln_gelu<<<NTOT / 64, 256, 0, stream>>>(
            h, mask, sep_w + i * FCH * 3, sep_b + i * FCH,
            n1_g + i * FCH, n1_b + i * FCH, y16, dils[i]);
        k3_pointwise_wmma<<<NTOT / 128, 256, 0, stream>>>(
            w16 + (size_t)i * FCH * FCH, pw_b + i * FCH,
            n2_g + i * FCH, n2_b + i * FCH, h, y16,
            (i == 2) ? y16 : (_Float16*)nullptr, mask);
    }

    k4_proj_rqs<<<NTOT / 128, 256, 0, stream>>>(wp16, proj_b, y16, x, mask, out, part);
    k5_logdet_reduce<<<1, 128, 0, stream>>>(part, logdet);
}
